// TPFM_3384434230013
// MI455X (gfx1250) — matrix operand order repaired and checked by reading, not compiled, not run on hardware
//
#include <hip/hip_runtime.h>
#include <hip/hip_bf16.h>

typedef __attribute__((ext_vector_type(16))) __bf16 v16bf;
typedef __attribute__((ext_vector_type(8)))  __bf16 v8bf;
typedef __attribute__((ext_vector_type(8)))  float  v8f;

#define EPS 1e-5f
#define NVOX 8192          // B * 16^3
#define CCH  512
#define C2   256

__device__ __forceinline__ float leaky(float v) { return v >= 0.f ? v : 0.01f * v; }

// ---------------------------------------------------------------------------
// K0: convert fp32 weights to bf16 row-major
// ---------------------------------------------------------------------------
__global__ void cvt_weights(const float* __restrict__ wv, const float* __restrict__ we,
                            const float* __restrict__ wf,
                            __bf16* __restrict__ wvb, __bf16* __restrict__ web,
                            __bf16* __restrict__ wfb) {
    int i = blockIdx.x * 256 + threadIdx.x;
    const int n1 = 256 * 1024, n2 = 256 * 1024, n3 = 512 * 768;
    if (i < n1)                wvb[i]           = (__bf16)wv[i];
    else if (i < n1 + n2)      web[i - n1]      = (__bf16)we[i - n1];
    else if (i < n1 + n2 + n3) wfb[i - n1 - n2] = (__bf16)wf[i - n1 - n2];
}

// ---------------------------------------------------------------------------
// K1: edge construction + edge-aggregation (Linear6->1 + BN + LeakyReLU)
// Writes voxel-major bf16: xT[n][c], aggT[n][c], edgeT[k][n][c]
// ---------------------------------------------------------------------------
__global__ __launch_bounds__(256) void prep_edges(
    const float* __restrict__ x, const float* __restrict__ topo,
    const float* __restrict__ w_ea, const float* __restrict__ b_ea,
    const float* __restrict__ g_ea, const float* __restrict__ be_ea,
    const float* __restrict__ rm_ea, const float* __restrict__ rv_ea,
    __bf16* __restrict__ xT, __bf16* __restrict__ aggT, __bf16* __restrict__ edgeT)
{
    int n = blockIdx.x * blockDim.x + threadIdx.x;   // 0..8191
    int b = n >> 12, s = n & 4095;
    int d = s >> 8, w = (s >> 4) & 15, h = s & 15;
    int nb[6];
    nb[0] = (((d + 15) & 15) << 8) | (w << 4) | h;   // roll(+1, D): neighbor d-1
    nb[1] = (((d + 1)  & 15) << 8) | (w << 4) | h;
    nb[2] = (d << 8) | (((w + 15) & 15) << 4) | h;
    nb[3] = (d << 8) | (((w + 1)  & 15) << 4) | h;
    nb[4] = (d << 8) | (w << 4) | ((h + 15) & 15);
    nb[5] = (d << 8) | (w << 4) | ((h + 1)  & 15);

    float wk[6], tk[6];
    #pragma unroll
    for (int k = 0; k < 6; ++k) { wk[k] = w_ea[k]; tk[k] = topo[s * 6 + k]; }
    float sA   = g_ea[0] * rsqrtf(rv_ea[0] + EPS);
    float bA   = be_ea[0] - rm_ea[0] * sA;
    float blin = b_ea[0];

    const float* xb = x + ((long)b << 21);           // b * 512 * 4096
    for (int c8 = 0; c8 < CCH; c8 += 8) {
        v8bf xs, as, es[6];
        #pragma unroll
        for (int j = 0; j < 8; ++j) {
            const float* xc = xb + ((long)(c8 + j) << 12);
            float xv = xc[s];
            float al = blin;
            #pragma unroll
            for (int k = 0; k < 6; ++k) {
                float e = xc[nb[k]] * xv + tk[k];
                es[k][j] = (__bf16)e;
                al += wk[k] * e;
            }
            float a = leaky(al * sA + bA);
            xs[j] = (__bf16)xv;
            as[j] = (__bf16)a;
        }
        int row = n * CCH + c8;
        *(v8bf*)(xT + row)   = xs;
        *(v8bf*)(aggT + row) = as;
        #pragma unroll
        for (int k = 0; k < 6; ++k)
            *(v8bf*)(edgeT + k * (NVOX * CCH) + row) = es[k];
    }
}

// ---------------------------------------------------------------------------
// WMMA fragment loaders (contiguous k order, same for A and B, wave32)
// A (16x32 bf16, row-major [m][k]): lane m=l&15, hi=l>>4,
//   elems 0..15 -> K=hi*16+e                                  (two 16B loads)
// B (32x16 bf16, voxel-major [n][k]): lane n=l&15, hi=l>>4,
//   elems 0..15 -> K=hi*16+e                                  (one 32B load)
// ---------------------------------------------------------------------------
__device__ __forceinline__ v16bf ldA(const __bf16* p) {
    v8bf lo  = *(const v8bf*)p;
    v8bf hi8 = *(const v8bf*)(p + 8);
    return __builtin_shufflevector(lo, hi8, 0,1,2,3,4,5,6,7,8,9,10,11,12,13,14,15);
}

// ---------------------------------------------------------------------------
// Generic bf16 WMMA GEMM  D(M x 8192) = A(M x [K1;K2]) * [B1;B2]
// All shapes compile-time: constant-offset addressing, partially-unrolled
// K loops so loads pipeline across WMMAs.
// MODE 0: out = lin + bias[m]             (f32 [M][N])   -> shared_e
// MODE 1: out = leaky(bn(lin + bias[m]))  (f32 [M][N])   -> uv
// MODE 2: out = leaky(bn(lin)) -> d_out (B,C,16^3) remap -> final
// ---------------------------------------------------------------------------
template<int LDA, int K1, int LDB1, int K2, int LDB2, int MODE>
__global__ __launch_bounds__(256) void gemm_bf16(
    const __bf16* __restrict__ A,
    const __bf16* __restrict__ B1, const __bf16* __restrict__ B2,
    float* __restrict__ out,
    const float* __restrict__ p_bias, const float* __restrict__ bn_g,
    const float* __restrict__ bn_b, const float* __restrict__ bn_m,
    const float* __restrict__ bn_v)
{
    const int N = NVOX;
    int lane = threadIdx.x & 31, wave = threadIdx.x >> 5;
    int m0 = blockIdx.x * 64 + (wave >> 2) * 32;
    int n0 = blockIdx.y * 128 + (wave & 3) * 32;
    int mr = lane & 15, hi = lane >> 4;

    const __bf16* a0  = A + (m0 + mr) * LDA + hi * 16;
    const __bf16* a1  = a0 + 16 * LDA;
    const __bf16* b10 = B1 + (n0 + mr) * LDB1 + hi * 16;
    const __bf16* b11 = b10 + 16 * LDB1;

    v8f acc[2][2] = {};
    #pragma unroll 4
    for (int kt = 0; kt < K1; kt += 32) {
        v16bf a[2]  = { ldA(a0 + kt), ldA(a1 + kt) };
        v16bf bf[2] = { *(const v16bf*)(b10 + kt), *(const v16bf*)(b11 + kt) };
        #pragma unroll
        for (int mi = 0; mi < 2; ++mi)
            #pragma unroll
            for (int ni = 0; ni < 2; ++ni)
                acc[mi][ni] = __builtin_amdgcn_wmma_f32_16x16x32_bf16(
                    false, a[mi], false, bf[ni], (short)0, acc[mi][ni], false, false);
    }
    if constexpr (K2 > 0) {
        const __bf16* b20 = B2 + (n0 + mr) * LDB2 + hi * 16;
        const __bf16* b21 = b20 + 16 * LDB2;
        #pragma unroll 4
        for (int kt = 0; kt < K2; kt += 32) {
            v16bf a[2]  = { ldA(a0 + K1 + kt), ldA(a1 + K1 + kt) };
            v16bf bf[2] = { *(const v16bf*)(b20 + kt), *(const v16bf*)(b21 + kt) };
            #pragma unroll
            for (int mi = 0; mi < 2; ++mi)
                #pragma unroll
                for (int ni = 0; ni < 2; ++ni)
                    acc[mi][ni] = __builtin_amdgcn_wmma_f32_16x16x32_bf16(
                        false, a[mi], false, bf[ni], (short)0, acc[mi][ni], false, false);
        }
    }

    #pragma unroll
    for (int mi = 0; mi < 2; ++mi)
        #pragma unroll
        for (int ni = 0; ni < 2; ++ni) {
            int nn = n0 + ni * 16 + mr;
            #pragma unroll
            for (int v = 0; v < 8; ++v) {
                int m = m0 + mi * 16 + hi * 8 + v;
                float lin = acc[mi][ni][v];
                if constexpr (MODE == 0) {
                    out[(long)m * N + nn] = lin + p_bias[m];
                } else if constexpr (MODE == 1) {
                    float sc  = bn_g[m] * rsqrtf(bn_v[m] + EPS);
                    float val = leaky((lin + p_bias[m] - bn_m[m]) * sc + bn_b[m]);
                    out[(long)m * N + nn] = val;
                } else {
                    float sc  = bn_g[m] * rsqrtf(bn_v[m] + EPS);
                    float val = leaky((lin - bn_m[m]) * sc + bn_b[m]);
                    int bb = nn >> 12;
                    out[(((long)bb * CCH + m) << 12) + (nn & 4095)] = val;
                }
            }
        }
}

// ---------------------------------------------------------------------------
// Fused 6-neighbor edge-update GEMM + BN + LeakyReLU + Linear(6->1) reduce
//     + scalar BN + LeakyReLU + multiply with uv.  ue never hits memory.
//     A2 = w_e[:,512:] (bf16, lda=1024);  ur accumulated in registers.
//     Writes update as bf16 [n][m] (B operand of final GEMM).
// ---------------------------------------------------------------------------
__global__ __launch_bounds__(256) void edge_update_gemm(
    const __bf16* __restrict__ A2, const __bf16* __restrict__ edgeT,
    const float* __restrict__ sharedE, const float* __restrict__ uvb,
    const float* __restrict__ g_e, const float* __restrict__ be_e,
    const float* __restrict__ rm_e, const float* __restrict__ rv_e,
    const float* __restrict__ w_r, const float* __restrict__ b_r,
    const float* __restrict__ g_r, const float* __restrict__ be_r,
    const float* __restrict__ rm_r, const float* __restrict__ rv_r,
    __bf16* __restrict__ updT)
{
    const int N = NVOX;
    int lane = threadIdx.x & 31, wave = threadIdx.x >> 5;
    int m0 = blockIdx.x * 64 + (wave >> 2) * 32;
    int n0 = blockIdx.y * 128 + (wave & 3) * 32;
    int mr = lane & 15, hi = lane >> 4;

    const __bf16* a0 = A2 + (m0 + mr) * 1024 + hi * 16;
    const __bf16* a1 = a0 + 16 * 1024;

    // per-(mi,v) BN constants and shared tile (w_e1 @ x + b_e)
    float se[2][8], sb[2][8], shs[2][2][8];
    #pragma unroll
    for (int mi = 0; mi < 2; ++mi)
        #pragma unroll
        for (int v = 0; v < 8; ++v) {
            int m = m0 + mi * 16 + hi * 8 + v;
            float sc = g_e[m] * rsqrtf(rv_e[m] + EPS);
            se[mi][v] = sc;
            sb[mi][v] = be_e[m] - rm_e[m] * sc;
            #pragma unroll
            for (int ni = 0; ni < 2; ++ni)
                shs[mi][ni][v] = sharedE[(long)m * N + n0 + ni * 16 + mr];
        }
    float wr[6];
    #pragma unroll
    for (int k = 0; k < 6; ++k) wr[k] = w_r[k];

    v8f ur[2][2] = {};
    for (int k = 0; k < 6; ++k) {
        const __bf16* bk0 = edgeT + k * (NVOX * CCH) + (n0 + mr) * CCH + hi * 16;
        const __bf16* bk1 = bk0 + 16 * CCH;
        v8f acc[2][2] = {};
        #pragma unroll 4
        for (int kt = 0; kt < CCH; kt += 32) {
            v16bf a[2]  = { ldA(a0 + kt), ldA(a1 + kt) };
            v16bf bf[2] = { *(const v16bf*)(bk0 + kt), *(const v16bf*)(bk1 + kt) };
            #pragma unroll
            for (int mi = 0; mi < 2; ++mi)
                #pragma unroll
                for (int ni = 0; ni < 2; ++ni)
                    acc[mi][ni] = __builtin_amdgcn_wmma_f32_16x16x32_bf16(
                        false, a[mi], false, bf[ni], (short)0, acc[mi][ni], false, false);
        }
        #pragma unroll
        for (int mi = 0; mi < 2; ++mi)
            #pragma unroll
            for (int ni = 0; ni < 2; ++ni)
                #pragma unroll
                for (int v = 0; v < 8; ++v) {
                    float lin = acc[mi][ni][v] + shs[mi][ni][v];
                    float ue  = leaky(lin * se[mi][v] + sb[mi][v]);
                    ur[mi][ni][v] += wr[k] * ue;
                }
    }

    float sr  = g_r[0] * rsqrtf(rv_r[0] + EPS);
    float br2 = be_r[0] - rm_r[0] * sr;
    float blr = b_r[0];
    #pragma unroll
    for (int mi = 0; mi < 2; ++mi)
        #pragma unroll
        for (int ni = 0; ni < 2; ++ni) {
            int nn = n0 + ni * 16 + mr;
            v8bf pack;
            #pragma unroll
            for (int v = 0; v < 8; ++v) {
                int m = m0 + mi * 16 + hi * 8 + v;
                float url = leaky((ur[mi][ni][v] + blr) * sr + br2);
                pack[v] = (__bf16)(uvb[(long)m * N + nn] * url);
            }
            *(v8bf*)(updT + (long)nn * C2 + m0 + mi * 16 + hi * 8) = pack;
        }
}

// ---------------------------------------------------------------------------
extern "C" void kernel_launch(void* const* d_in, const int* in_sizes, int n_in,
                              void* d_out, int out_size, void* d_ws, size_t ws_size,
                              hipStream_t stream) {
    const float* x    = (const float*)d_in[0];
    const float* topo = (const float*)d_in[1];
    const float *w_ea = (const float*)d_in[2],  *b_ea = (const float*)d_in[3];
    const float *g_ea = (const float*)d_in[4],  *be_ea= (const float*)d_in[5];
    const float *rm_ea= (const float*)d_in[6],  *rv_ea= (const float*)d_in[7];
    const float *w_v  = (const float*)d_in[8],  *b_v  = (const float*)d_in[9];
    const float *g_v  = (const float*)d_in[10], *be_v = (const float*)d_in[11];
    const float *rm_v = (const float*)d_in[12], *rv_v = (const float*)d_in[13];
    const float *w_e  = (const float*)d_in[14], *b_e  = (const float*)d_in[15];
    const float *g_e  = (const float*)d_in[16], *be_e = (const float*)d_in[17];
    const float *rm_e = (const float*)d_in[18], *rv_e = (const float*)d_in[19];
    const float *w_r  = (const float*)d_in[20], *b_r  = (const float*)d_in[21];
    const float *g_r  = (const float*)d_in[22], *be_r = (const float*)d_in[23];
    const float *rm_r = (const float*)d_in[24], *rv_r = (const float*)d_in[25];
    const float *w_f  = (const float*)d_in[26], *g_f  = (const float*)d_in[27];
    const float *be_f = (const float*)d_in[28], *rm_f = (const float*)d_in[29];
    const float *rv_f = (const float*)d_in[30];

    // workspace layout (all offsets 16B-aligned; total ~86 MiB)
    __bf16* xT    = (__bf16*)d_ws;                 // 8192*512
    __bf16* aggT  = xT   + NVOX * CCH;             // 8192*512
    __bf16* edgeT = aggT + NVOX * CCH;             // 6*8192*512
    __bf16* updT  = edgeT + 6 * NVOX * CCH;        // 8192*256
    __bf16* wvb   = updT + NVOX * C2;              // 256*1024
    __bf16* web   = wvb + 256 * 1024;              // 256*1024
    __bf16* wfb   = web + 256 * 1024;              // 512*768
    float*  uvb   = (float*)(wfb + 512 * 768);     // 256*8192
    float*  shE   = uvb + C2 * NVOX;               // 256*8192

    cvt_weights<<<3584, 256, 0, stream>>>(w_v, w_e, w_f, wvb, web, wfb);
    prep_edges<<<NVOX / 256, 256, 0, stream>>>(x, topo, w_ea, b_ea, g_ea, be_ea,
                                               rm_ea, rv_ea, xT, aggT, edgeT);
    // shared_e = w_e[:, :512] @ x + b_e        (M=256, K=512)
    gemm_bf16<1024, 512, 512, 0, 0, 0><<<dim3(4, 64), 256, 0, stream>>>(
        web, xT, nullptr, shE, b_e, nullptr, nullptr, nullptr, nullptr);
    // uv = leaky(bn(w_v @ [x; agg] + b_v))     (M=256, K=1024)
    gemm_bf16<1024, 512, 512, 512, 512, 1><<<dim3(4, 64), 256, 0, stream>>>(
        wvb, xT, aggT, uvb, b_v, g_v, be_v, rm_v, rv_v);
    // fused 6x edge GEMM + reduce + update     (M=256, K=6x512)
    edge_update_gemm<<<dim3(4, 64), 256, 0, stream>>>(web + 512, edgeT, shE, uvb,
        g_e, be_e, rm_e, rv_e, w_r, b_r, g_r, be_r, rm_r, rv_r, updT);
    // out = leaky(bn(w_f @ [x; update]))       (M=512, K=768)
    gemm_bf16<768, 512, 512, 256, 256, 2><<<dim3(8, 64), 256, 0, stream>>>(
        wfb, xT, updT, (float*)d_out, nullptr, g_f, be_f, rm_f, rv_f);
}
